// VectorQuantizer_16338055594251
// MI455X (gfx1250) — matrix operand order repaired and checked by reading, not compiled, not run on hardware
//
#include <hip/hip_runtime.h>
#include <hip/hip_bf16.h>
#include <math.h>

typedef __attribute__((ext_vector_type(16))) _Float16 v16h;
typedef __attribute__((ext_vector_type(8)))  _Float16 v8h;
typedef __attribute__((ext_vector_type(8)))  float    v8f;

#define N_ROWS   (16 * 16384)     // 262144 flattened z rows
#define DDIM     64
#define SUBN     256              // sub-codebook size
#define T_PER_B  16384
#define ROWS_PER_BLOCK 128
#define NBLOCKS  (N_ROWS / ROWS_PER_BLOCK)   // 2048

// d_out layout (flat, float32): z_q_out [16,64,16384] | loss | perplexity | idx [262144]
#define LOSS_OFF 16777216
#define PERP_OFF 16777217
#define IDX_OFF  16777218

// d_ws layout: int counts[256] ; float partial_loss[NBLOCKS]

__global__ void vq_init_kernel(int* __restrict__ counts) {
    counts[threadIdx.x] = 0;
}

__global__ __launch_bounds__(256)
void vq_main_kernel(const float* __restrict__ z,
                    const float* __restrict__ one_hot,
                    const float* __restrict__ W,
                    float* __restrict__ out,
                    int* __restrict__ counts,
                    float* __restrict__ partial)
{
    // LDS: f16 codebook (padded stride 72 halves -> conflict-free ds loads),
    // transposed f32 ||e||^2 (sCEt[m][jt]), histogram, per-wave argmin, loss scratch.
    __shared__ _Float16 sEh[SUBN * 72];
    __shared__ float    sCEt[SUBN];
    __shared__ int      sHist[SUBN];
    __shared__ int      sIdx[8][16];
    __shared__ float    sLoss[256];
    __shared__ int      sPos;

    const int tid = threadIdx.x;

    if (tid == 0) {
        int p = 0; float bv = one_hot[0];
        #pragma unroll
        for (int i = 1; i < 7; ++i) { float v = one_hot[i]; if (v > bv) { bv = v; p = i; } }
        sPos = p;
    }
    sHist[tid] = 0;
    __syncthreads();

    const int pos = sPos;
    const float* __restrict__ E = W + (size_t)pos * SUBN * DDIM;

    // Stage E into LDS as f16 (row stride 72 halves).
    for (int i = tid; i < SUBN * DDIM; i += 256) {
        float v = E[i];
        sEh[(i >> 6) * 72 + (i & 63)] = (_Float16)v;
    }
    // Exact fp32 ||e_j||^2 for code row tid, stored transposed: sCEt[(j%16)*16 + (j/16)]
    {
        const float* er = E + (size_t)tid * DDIM;
        float s = 0.f;
        #pragma unroll 8
        for (int d = 0; d < DDIM; ++d) { float v = er[d]; s = fmaf(v, v, s); }
        sCEt[(tid & 15) * 16 + (tid >> 4)] = s;
    }
    __syncthreads();

    const int wave   = tid >> 5;       // 8 waves of 32 (wave32)
    const int lane   = tid & 31;
    const int half16 = lane >> 4;      // 0: lanes 0-15, 1: lanes 16-31
    const int m      = lane & 15;
    const int rowbase = blockIdx.x * ROWS_PER_BLOCK + wave * 16;
    const int row     = rowbase + m;   // this lane's z row
    const int c0      = half16 * 8;    // K sub-offset per documented A layout

    // Load this lane's 32 floats of its z row (K chunks c0.., 16+c0.., 32+c0.., 48+c0..)
    union ZF { float4 q[8]; float s[32]; } zf;
    {
        const float4* zp = (const float4*)(z + (size_t)row * DDIM + c0);
        zf.q[0] = zp[0];  zf.q[1] = zp[1];    // K = c0 .. c0+7
        zf.q[2] = zp[4];  zf.q[3] = zp[5];    // K = 16+c0 ..
        zf.q[4] = zp[8];  zf.q[5] = zp[9];    // K = 32+c0 ..
        zf.q[6] = zp[12]; zf.q[7] = zp[13];   // K = 48+c0 ..
    }

    // A fragments per 16-bit A 16x32 layout:
    // lane<16: halves 0-7 -> K 0-7, halves 8-15 -> K 16-23 ; lane>=16: +8
    union V16 { v16h v; v8h h8[2]; _Float16 h[16]; };
    V16 a0, a1;
    #pragma unroll
    for (int k = 0; k < 16; ++k) {
        a0.h[k] = (_Float16)zf.s[k];        // K-lo half (0..31 region)
        a1.h[k] = (_Float16)zf.s[16 + k];   // K-hi half (32..63 region)
    }

    // Preload this lane's 16 ||e||^2 values (one per column tile) in 4 b128 loads.
    union CE { float4 q[4]; float s[16]; } cev;
    {
        const float4* cp = (const float4*)(sCEt + m * 16);
        cev.q[0] = cp[0]; cev.q[1] = cp[1]; cev.q[2] = cp[2]; cev.q[3] = cp[3];
    }

    float best[8];
    int   bidx[8];
    #pragma unroll
    for (int r = 0; r < 8; ++r) { best[r] = 3.0e38f; bidx[r] = 0; }

    auto loadB = [&](V16& B0, V16& B1, int jt) {
        const _Float16* bp = sEh + (size_t)(jt * 16 + m) * 72 + half16 * 8;
        B0.h8[0] = *(const v8h*)(bp);      B0.h8[1] = *(const v8h*)(bp + 16);
        B1.h8[0] = *(const v8h*)(bp + 32); B1.h8[1] = *(const v8h*)(bp + 48);
    };
    auto upd = [&](int jt, const v8f& c) {
        const int   j  = jt * 16 + m;
        const float ce = cev.s[jt];
        #pragma unroll
        for (int r = 0; r < 8; ++r) {
            float d = fmaf(-2.0f, c[r], ce);   // ||e||^2 - 2 z.e
            if (d < best[r]) { best[r] = d; bidx[r] = j; }  // strict < keeps lowest j
        }
    };

    // Software-pipelined: B double-buffered one tile ahead; argmin of tile jt-1
    // co-executes under tile jt's WMMAs (hides WMMA->VALU hazard gap).
    V16 b0[2], b1[2];
    loadB(b0[0], b1[0], 0);
    loadB(b0[1], b1[1], 1);
    v8f cprev = {};
    cprev = __builtin_amdgcn_wmma_f32_16x16x32_f16(false, a0.v, false, b0[0].v,
                                                   (short)0, cprev, false, false);
    cprev = __builtin_amdgcn_wmma_f32_16x16x32_f16(false, a1.v, false, b1[0].v,
                                                   (short)0, cprev, false, false);
    #pragma unroll
    for (int jt = 1; jt < 16; ++jt) {
        const int cur = jt & 1;
        const int nxt = cur ^ 1;
        if (jt < 15) loadB(b0[nxt], b1[nxt], jt + 1);
        v8f ccur = {};
        ccur = __builtin_amdgcn_wmma_f32_16x16x32_f16(false, a0.v, false, b0[cur].v,
                                                      (short)0, ccur, false, false);
        ccur = __builtin_amdgcn_wmma_f32_16x16x32_f16(false, a1.v, false, b1[cur].v,
                                                      (short)0, ccur, false, false);
        upd(jt - 1, cprev);
        cprev = ccur;
    }
    upd(15, cprev);

    // Butterfly argmin across the 16 columns (stays within each half-wave).
    #pragma unroll
    for (int off = 8; off >= 1; off >>= 1) {
        #pragma unroll
        for (int r = 0; r < 8; ++r) {
            float ob = __shfl_xor(best[r], off, 32);
            int   oi = __shfl_xor(bidx[r], off, 32);
            if (ob < best[r] || (ob == best[r] && oi < bidx[r])) { best[r] = ob; bidx[r] = oi; }
        }
    }
    if (m == 0) {     // lanes 0 (rows 0-7) and 16 (rows 8-15)
        #pragma unroll
        for (int r = 0; r < 8; ++r) sIdx[wave][half16 * 8 + r] = bidx[r];
    }
    __syncthreads();

    const int myIdx = sIdx[wave][m];
    if (lane < 16) {
        out[IDX_OFF + row] = (float)myIdx;       // idx output (float-cast)
        atomicAdd(&sHist[myIdx], 1);
    }

    // Transposed write-out: out[b, d, t] = E[idx[n]][d], n = b*16384 + t.
    // Lane handles row t=rowbase+m, its own 32 d's; each i-step is 64B contiguous per half-wave.
    const size_t bb = (size_t)(row / T_PER_B);
    const size_t t  = (size_t)(row % T_PER_B);
    float* ob = out + bb * (size_t)(DDIM * T_PER_B) + t;
    const float* eg = E + (size_t)myIdx * DDIM;
    float lsum = 0.f;
    #pragma unroll
    for (int c = 0; c < 4; ++c) {
        #pragma unroll
        for (int i = 0; i < 8; ++i) {
            const int d = c * 16 + c0 + i;
            const float e = eg[d];
            ob[(size_t)d * T_PER_B] = e;
            const float df = e - zf.s[c * 8 + i];
            lsum = fmaf(df, df, lsum);
        }
    }

    // Block loss reduction + histogram flush.
    sLoss[tid] = lsum;
    __syncthreads();
    #pragma unroll
    for (int s2 = 128; s2 > 0; s2 >>= 1) {
        if (tid < s2) sLoss[tid] += sLoss[tid + s2];
        __syncthreads();
    }
    if (tid == 0) partial[blockIdx.x] = sLoss[0];
    const int h = sHist[tid];
    if (h) atomicAdd(&counts[tid], h);
}

__global__ __launch_bounds__(256)
void vq_final_kernel(const int* __restrict__ counts,
                     const float* __restrict__ partial,
                     float* __restrict__ out)
{
    __shared__ float sA[256], sB[256];
    const int tid = threadIdx.x;
    float ls = 0.f;
    for (int i = tid; i < NBLOCKS; i += 256) ls += partial[i];
    const float em  = (float)counts[tid] / (float)N_ROWS;
    const float ent = em * logf(em + 1e-10f);
    sA[tid] = ls; sB[tid] = ent;
    __syncthreads();
    #pragma unroll
    for (int s = 128; s > 0; s >>= 1) {
        if (tid < s) { sA[tid] += sA[tid + s]; sB[tid] += sB[tid + s]; }
        __syncthreads();
    }
    if (tid == 0) {
        const float mse = sA[0] / (float)((size_t)N_ROWS * DDIM);
        out[LOSS_OFF] = 1.25f * mse;     // BETA*mse + mse, forward value
        out[PERP_OFF] = expf(-sB[0]);
    }
}

extern "C" void kernel_launch(void* const* d_in, const int* in_sizes, int n_in,
                              void* d_out, int out_size, void* d_ws, size_t ws_size,
                              hipStream_t stream)
{
    const float* z       = (const float*)d_in[0];   // [16,16384,64] f32
    const float* one_hot = (const float*)d_in[1];   // [7] f32
    const float* W       = (const float*)d_in[2];   // [1792,64] f32
    float* out = (float*)d_out;
    int*   counts  = (int*)d_ws;
    float* partial = (float*)d_ws + SUBN;

    vq_init_kernel<<<1, SUBN, 0, stream>>>(counts);
    vq_main_kernel<<<NBLOCKS, 256, 0, stream>>>(z, one_hot, W, out, counts, partial);
    vq_final_kernel<<<1, 256, 0, stream>>>(counts, partial, out);
}
